// MambaBlock_19172734009546
// MI455X (gfx1250) — matrix operand order repaired and checked by reading, not compiled, not run on hardware
//
#include <hip/hip_runtime.h>
#include <hip/hip_bf16.h>
#include <stdint.h>

// ---------------- types ----------------
typedef __bf16 bf16_t;
typedef __attribute__((ext_vector_type(16))) __bf16 v16bf;
typedef __attribute__((ext_vector_type(8)))  __bf16 v8bf;
typedef __attribute__((ext_vector_type(8)))  float  v8f;

// ---------------- problem sizes ----------------
#define D_MODEL 1024
#define D_STATE 16
#define D_CONVW 4
#define D_INNER 2048
#define DT_RANK 64
#define BB      4
#define LL      2048
#define NROW    (BB * LL)                  // 8192 tokens
#define XDBL_W  (DT_RANK + 2 * D_STATE)    // 96

// ---------------- helpers ----------------
static __device__ __forceinline__ bf16_t f2bf(float f) {
    union { float f; uint32_t u; } v; v.f = f;
    uint32_t u = v.u;
    uint32_t r = (u + 0x7FFFu + ((u >> 16) & 1u)) >> 16;   // RNE
    union { uint16_t s; bf16_t b; } o; o.s = (uint16_t)r;
    return o.b;
}

static __device__ __forceinline__ v8f v8f_zero() {
    v8f r;
#pragma unroll
    for (int i = 0; i < 8; ++i) r[i] = 0.0f;
    return r;
}

static __device__ __forceinline__ v16bf combine16(v8bf lo, v8bf hi) {
    v16bf r;
#pragma unroll
    for (int e = 0; e < 8; ++e) { r[e] = lo[e]; r[e + 8] = hi[e]; }
    return r;
}

// ---------------- 1) LayerNorm -> bf16 ----------------
__global__ __launch_bounds__(256) void ln_kernel(const float* __restrict__ x,
                                                 const float* __restrict__ w,
                                                 const float* __restrict__ b,
                                                 bf16_t* __restrict__ xn) {
    int row = blockIdx.x;
    int tid = threadIdx.x;
    const float4* x4 = reinterpret_cast<const float4*>(x + (size_t)row * D_MODEL);
    float4 v = x4[tid];
    float s  = v.x + v.y + v.z + v.w;
    float ss = v.x * v.x + v.y * v.y + v.z * v.z + v.w * v.w;

    __shared__ float red[256];
    red[tid] = s; __syncthreads();
#pragma unroll
    for (int o = 128; o > 0; o >>= 1) { if (tid < o) red[tid] += red[tid + o]; __syncthreads(); }
    float mean = red[0] * (1.0f / D_MODEL);
    __syncthreads();
    red[tid] = ss; __syncthreads();
#pragma unroll
    for (int o = 128; o > 0; o >>= 1) { if (tid < o) red[tid] += red[tid + o]; __syncthreads(); }
    float var  = red[0] * (1.0f / D_MODEL) - mean * mean;
    float rstd = rsqrtf(var + 1e-5f);

    int c = tid * 4;
    bf16_t* o = xn + (size_t)row * D_MODEL + c;
    o[0] = f2bf((v.x - mean) * rstd * w[c + 0] + b[c + 0]);
    o[1] = f2bf((v.y - mean) * rstd * w[c + 1] + b[c + 1]);
    o[2] = f2bf((v.z - mean) * rstd * w[c + 2] + b[c + 2]);
    o[3] = f2bf((v.w - mean) * rstd * w[c + 3] + b[c + 3]);
}

// ---------------- 2) fp32 -> bf16 convert ----------------
__global__ __launch_bounds__(256) void f2bf_kernel(const float* __restrict__ src,
                                                   bf16_t* __restrict__ dst, int n) {
    int i = blockIdx.x * 256 + threadIdx.x;
    if (i < n) dst[i] = f2bf(src[i]);
}

// compact dt columns (first 64 of 96-wide x_dbl rows) to bf16
__global__ __launch_bounds__(256) void dt_compact_kernel(const float* __restrict__ xdbl,
                                                         bf16_t* __restrict__ dtbf) {
    int i = blockIdx.x * 256 + threadIdx.x;
    if (i < NROW * DT_RANK) {
        int r = i / DT_RANK, c = i % DT_RANK;
        dtbf[i] = f2bf(xdbl[(size_t)r * XDBL_W + c]);
    }
}

// ---------------- 3) generic bf16 WMMA GEMM ----------------
// C[M,N](f32) = A[M,K](bf16, row major) * W[N,K](bf16, row major)^T  (+ residual)
// Both tiles staged row-major in LDS via global_load_async_to_lds_b128 (ASYNCcnt).
// ISA layouts (7.12.2):
//   A frag: lane<16 -> M=lane,  elems[0..7]=K(koff..koff+7), [8..15]=K(koff+16..+23), koff = 0/8
//   B frag: lane&15 -> N col,   elems[0..7]=K(koffB..koffB+7), [8..15]=K(koffB+16..+23), koffB = 0/8
#define BM 128
#define BN 128
#define BK 32
#define ALD 40                       // padded tile leading dim (80B rows, 16B aligned)
#define WOFF (BM * ALD)              // Ws base (elements) inside dynamic LDS
#define GEMM_SMEM ((size_t)(2 * BM * ALD * 2))   // bytes

__global__ __launch_bounds__(256) void gemm_bf16_wmma(const bf16_t* __restrict__ A,
                                                      const bf16_t* __restrict__ W,
                                                      float* __restrict__ C,
                                                      const float* __restrict__ R,
                                                      int M, int N, int K) {
    extern __shared__ __align__(16) bf16_t smem[];
    bf16_t* As = smem;               // [BM][ALD]
    bf16_t* Ws = smem + WOFF;        // [BN][ALD]

    int tid  = threadIdx.x;
    int lane = tid & 31;
    int wave = tid >> 5;             // 0..7
    int wm   = wave & 3;             // 4 waves along M (32 rows each)
    int wn   = wave >> 2;            // 2 waves along N (64 cols each)
    int bM   = blockIdx.y * BM;
    int bN   = blockIdx.x * BN;

    v8f acc[2][4];
#pragma unroll
    for (int i = 0; i < 2; ++i)
#pragma unroll
        for (int j = 0; j < 4; ++j) acc[i][j] = v8f_zero();

    // staging assignment: 2 threads per tile row, 16 bf16 (32B) each
    int rowT = tid >> 1;                       // 0..127
    int kh   = (tid & 1) * 16;                 // 0 or 16
    const bf16_t* gA = A + (size_t)(bM + rowT) * K + kh;
    int nW  = bN + rowT;
    int nWc = (nW < N) ? nW : (N - 1);         // clamp (cols >= N never stored)
    const bf16_t* gW = W + (size_t)nWc * K + kh;
    unsigned aoff = (unsigned)((rowT * ALD + kh) * 2);          // LDS byte offset (A)
    unsigned woff = (unsigned)((WOFF + rowT * ALD + kh) * 2);   // LDS byte offset (W)

    for (int k0 = 0; k0 < K; k0 += BK) {
        __syncthreads();                       // previous iteration's LDS reads done

        // async global -> LDS tile staging (inst offset applies to both addresses)
        const bf16_t* pa = gA + k0;
        const bf16_t* pw = gW + k0;
        asm volatile("global_load_async_to_lds_b128 %0, %1, off"
                     :: "v"(aoff), "v"(pa) : "memory");
        asm volatile("global_load_async_to_lds_b128 %0, %1, off offset:16"
                     :: "v"(aoff), "v"(pa) : "memory");
        asm volatile("global_load_async_to_lds_b128 %0, %1, off"
                     :: "v"(woff), "v"(pw) : "memory");
        asm volatile("global_load_async_to_lds_b128 %0, %1, off offset:16"
                     :: "v"(woff), "v"(pw) : "memory");
        if (k0 + BK < K) {                     // hint next K tile -> global_prefetch_b8
            __builtin_prefetch(pa + BK, 0, 1);
            __builtin_prefetch(pw + BK, 0, 1);
        }
        asm volatile("s_wait_asynccnt 0x0" ::: "memory");
        __syncthreads();                       // tile visible to all waves

        // ---- A fragments
        int koffA = (lane >> 4) * 8;
        v16bf af[2];
#pragma unroll
        for (int i = 0; i < 2; ++i) {
            int rA = wm * 32 + i * 16 + (lane & 15);
            v8bf lo = *reinterpret_cast<const v8bf*>(&As[rA * ALD + koffA]);
            v8bf hi = *reinterpret_cast<const v8bf*>(&As[rA * ALD + koffA + 16]);
            af[i] = combine16(lo, hi);
        }
        // ---- B fragments (row-major W, same K order as A)
        int koffB = (lane >> 4) * 8;
        v16bf bfar[4];
#pragma unroll
        for (int j = 0; j < 4; ++j) {
            int nl = wn * 64 + j * 16 + (lane & 15);
            v8bf lo = *reinterpret_cast<const v8bf*>(&Ws[nl * ALD + koffB]);
            v8bf hi = *reinterpret_cast<const v8bf*>(&Ws[nl * ALD + koffB + 16]);
            bfar[j] = combine16(lo, hi);
        }
        // ---- 8 WMMAs (32x64 per wave per K-step)
#pragma unroll
        for (int i = 0; i < 2; ++i)
#pragma unroll
            for (int j = 0; j < 4; ++j)
                acc[i][j] = __builtin_amdgcn_wmma_f32_16x16x32_bf16(
                    false, af[i], false, bfar[j], (short)0, acc[i][j], false, false);
    }

    // ---- epilogue: C/D layout — lane<16: M=r, lane>=16: M=r+8 ; N = lane&15
    int mrow = (lane < 16) ? 0 : 8;
    int ncol = lane & 15;
#pragma unroll
    for (int i = 0; i < 2; ++i)
#pragma unroll
        for (int j = 0; j < 4; ++j) {
            int gn = bN + wn * 64 + j * 16 + ncol;
            if (gn < N) {
#pragma unroll
                for (int r = 0; r < 8; ++r) {
                    int gm = bM + wm * 32 + i * 16 + mrow + r;
                    size_t idx = (size_t)gm * N + gn;
                    float val = acc[i][j][r];
                    if (R) val += R[idx];
                    C[idx] = val;
                }
            }
        }
}

// ---------------- 4) causal depthwise conv + SiLU ----------------
__global__ __launch_bounds__(256) void conv_silu_kernel(const float* __restrict__ uz,
                                                        const float* __restrict__ cw,
                                                        const float* __restrict__ cb,
                                                        float* __restrict__ uc,
                                                        bf16_t* __restrict__ ucbf) {
    size_t i  = (size_t)blockIdx.x * 256 + threadIdx.x;   // over NROW*D_INNER
    int    d  = (int)(i & (D_INNER - 1));
    size_t bt = i >> 11;                                   // token row (b*L + t)
    int    t  = (int)(bt & (LL - 1));
    float acc = cb[d];
#pragma unroll
    for (int j = 0; j < D_CONVW; ++j) {
        int tt = t - 3 + j;
        if (tt >= 0) acc += cw[d * D_CONVW + j] * uz[(bt - 3 + j) * (size_t)(2 * D_INNER) + d];
    }
    float r = acc / (1.0f + __expf(-acc));   // SiLU
    uc[i]   = r;
    ucbf[i] = f2bf(r);
}

// ---------------- 5) selective scan: 16 lanes per channel (one state each) -------
// block = 256 threads = 16 channels; y staged in LDS, flushed coalesced every 16 t.
__global__ __launch_bounds__(256) void scan_kernel(const float* __restrict__ draw,   // [NROW, D_INNER]
                                                   const float* __restrict__ uconv,  // [NROW, D_INNER]
                                                   const float* __restrict__ uz,     // [NROW, 2*D_INNER]
                                                   const float* __restrict__ xdbl,   // [NROW, 96]
                                                   const float* __restrict__ dtb,    // [D_INNER]
                                                   const float* __restrict__ Alog,   // [D_INNER, 16]
                                                   const float* __restrict__ Dp,     // [D_INNER]
                                                   bf16_t* __restrict__ ybf) {       // [NROW, D_INNER]
    __shared__ bf16_t ybuf[16 * 16];          // [t-sub][channel]
    int tid = threadIdx.x;
    int ch  = tid >> 4;                        // 0..15 channel in block
    int s   = tid & 15;                        // state index
    int cidx = blockIdx.x * 16 + ch;           // global channel 0..8191
    int b = cidx >> 11;
    int d = cidx & (D_INNER - 1);

    float As   = -__expf(Alog[d * D_STATE + s]);
    float h    = 0.0f;
    float bias = dtb[d];
    float Dd   = Dp[d];

    for (int t0 = 0; t0 < LL; t0 += 16) {
        for (int i = 0; i < 16; ++i) {
            size_t row = (size_t)b * LL + (t0 + i);
            float xr = draw[row * D_INNER + d] + bias;           // broadcast across 16 lanes
            float delta = (xr > 20.0f) ? xr : log1pf(__expf(xr)); // softplus
            float u  = uconv[row * D_INNER + d];
            float Bt = xdbl[row * XDBL_W + DT_RANK + s];          // coalesced over lanes
            float Ct = xdbl[row * XDBL_W + DT_RANK + D_STATE + s];

            h = __expf(delta * As) * h + (delta * u) * Bt;
            float p = h * Ct;
            p += __shfl_xor(p, 1);
            p += __shfl_xor(p, 2);
            p += __shfl_xor(p, 4);
            p += __shfl_xor(p, 8);                                // sum over 16 states
            if (s == 0) {
                float zv = uz[row * (size_t)(2 * D_INNER) + D_INNER + d];
                float zg = zv / (1.0f + __expf(-zv));             // SiLU(z)
                ybuf[i * 16 + ch] = f2bf((p + Dd * u) * zg);
            }
        }
        __syncthreads();
        {   // coalesced flush of 16 timesteps x 16 channels
            int i2 = tid >> 4, c2 = tid & 15;
            int cix2 = blockIdx.x * 16 + c2;
            int b2 = cix2 >> 11, d2 = cix2 & (D_INNER - 1);
            ybf[((size_t)b2 * LL + t0 + i2) * D_INNER + d2] = ybuf[i2 * 16 + c2];
        }
        __syncthreads();
    }
}

// ---------------- launcher ----------------
extern "C" void kernel_launch(void* const* d_in, const int* in_sizes, int n_in,
                              void* d_out, int out_size, void* d_ws, size_t ws_size,
                              hipStream_t stream) {
    (void)in_sizes; (void)n_in; (void)out_size; (void)ws_size;
    const float* x        = (const float*)d_in[0];
    const float* norm_w   = (const float*)d_in[1];
    const float* norm_b   = (const float*)d_in[2];
    const float* in_proj  = (const float*)d_in[3];   // [4096,1024]
    const float* conv_w   = (const float*)d_in[4];   // [2048,1,4]
    const float* conv_b   = (const float*)d_in[5];
    const float* x_proj   = (const float*)d_in[6];   // [96,2048]
    const float* dt_proj  = (const float*)d_in[7];   // [2048,64]
    const float* dt_bias  = (const float*)d_in[8];
    const float* A_log    = (const float*)d_in[9];   // [2048,16]
    const float* Dp       = (const float*)d_in[10];
    const float* out_proj = (const float*)d_in[11];  // [1024,2048]
    float* out = (float*)d_out;

    // ---- workspace carve-up (bytes) ----
    char* base = (char*)d_ws;
    size_t off = 0;
    auto carve = [&](size_t bytes) { char* p = base + off; off += (bytes + 255) & ~(size_t)255; return p; };
    bf16_t* xn    = (bf16_t*)carve((size_t)NROW * D_MODEL * 2);
    bf16_t* wA    = (bf16_t*)carve((size_t)2 * D_INNER * D_MODEL * 2);
    bf16_t* wX    = (bf16_t*)carve((size_t)XDBL_W * D_INNER * 2);
    bf16_t* wD    = (bf16_t*)carve((size_t)D_INNER * DT_RANK * 2);
    bf16_t* wO    = (bf16_t*)carve((size_t)D_MODEL * D_INNER * 2);
    float*  uzbuf = (float*) carve((size_t)NROW * 2 * D_INNER * 4);
    float*  uconv = (float*) carve((size_t)NROW * D_INNER * 4);
    bf16_t* ucbf  = (bf16_t*)carve((size_t)NROW * D_INNER * 2);
    float*  xdbl  = (float*) carve((size_t)NROW * XDBL_W * 4);
    bf16_t* dtbf  = (bf16_t*)carve((size_t)NROW * DT_RANK * 2);
    float*  draw  = (float*) carve((size_t)NROW * D_INNER * 4);
    bf16_t* ybf   = (bf16_t*)carve((size_t)NROW * D_INNER * 2);

    // 1) LayerNorm -> bf16
    ln_kernel<<<NROW, 256, 0, stream>>>(x, norm_w, norm_b, xn);

    // 2) weight conversions -> bf16
    {
        int n1 = 2 * D_INNER * D_MODEL;
        f2bf_kernel<<<(n1 + 255) / 256, 256, 0, stream>>>(in_proj, wA, n1);
        int n2 = XDBL_W * D_INNER;
        f2bf_kernel<<<(n2 + 255) / 256, 256, 0, stream>>>(x_proj, wX, n2);
        int n3 = D_INNER * DT_RANK;
        f2bf_kernel<<<(n3 + 255) / 256, 256, 0, stream>>>(dt_proj, wD, n3);
        int n4 = D_MODEL * D_INNER;
        f2bf_kernel<<<(n4 + 255) / 256, 256, 0, stream>>>(out_proj, wO, n4);
    }

    dim3 blk(256);
    // 3) in_proj GEMM: [8192,1024] x [4096,1024]^T -> uz [8192,4096]
    {
        dim3 grid((2 * D_INNER + BN - 1) / BN, NROW / BM);
        gemm_bf16_wmma<<<grid, blk, GEMM_SMEM, stream>>>(xn, wA, uzbuf, nullptr,
                                                         NROW, 2 * D_INNER, D_MODEL);
    }
    // 4) conv + SiLU on u half -> uconv (f32 + bf16)
    {
        size_t total = (size_t)NROW * D_INNER;
        conv_silu_kernel<<<(unsigned)(total / 256), blk, 0, stream>>>(uzbuf, conv_w, conv_b,
                                                                      uconv, ucbf);
    }
    // 5) x_proj GEMM: [8192,2048] x [96,2048]^T -> xdbl [8192,96]
    {
        dim3 grid((XDBL_W + BN - 1) / BN, NROW / BM);
        gemm_bf16_wmma<<<grid, blk, GEMM_SMEM, stream>>>(ucbf, wX, xdbl, nullptr,
                                                         NROW, XDBL_W, D_INNER);
    }
    // 6) compact dt columns -> bf16
    {
        int n = NROW * DT_RANK;
        dt_compact_kernel<<<(n + 255) / 256, 256, 0, stream>>>(xdbl, dtbf);
    }
    // 7) dt_proj GEMM: [8192,64] x [2048,64]^T -> draw [8192,2048]
    {
        dim3 grid((D_INNER + BN - 1) / BN, NROW / BM);
        gemm_bf16_wmma<<<grid, blk, GEMM_SMEM, stream>>>(dtbf, wD, draw, nullptr,
                                                         NROW, D_INNER, DT_RANK);
    }
    // 8) selective scan (+softplus, +SiLU(z) gate) -> ybf
    scan_kernel<<<(BB * D_INNER) / 16, blk, 0, stream>>>(draw, uconv, uzbuf, xdbl,
                                                         dt_bias, A_log, Dp, ybf);
    // 9) out_proj GEMM + residual: [8192,2048] x [1024,2048]^T + x -> out
    {
        dim3 grid((D_MODEL + BN - 1) / BN, NROW / BM);
        gemm_bf16_wmma<<<grid, blk, GEMM_SMEM, stream>>>(ybf, wO, out, x,
                                                         NROW, D_MODEL, D_INNER);
    }
}
